// LSTMLayer_66915590472020
// MI455X (gfx1250) — matrix operand order repaired and checked by reading, not compiled, not run on hardware
//
#include <hip/hip_runtime.h>
#include <hip/hip_bf16.h>

// ---------------------------------------------------------------------------
// LSTM layer for MI455X (gfx1250): persistent wave32 WMMA-bf16 kernel.
//   SEQ=512, B=64, IN=HS=1024, GATES=4096
// Each wave: 1 M-tile x 2 N-tiles (shared A fragment, 2 independent WMMA
// accumulator chains -> 3 b128 loads per WMMA instead of 4).
// ---------------------------------------------------------------------------

typedef __attribute__((ext_vector_type(16))) __bf16 v16bf;
typedef __attribute__((ext_vector_type(8)))  float  v8f;

namespace {
constexpr int kSeq    = 512;
constexpr int kBatch  = 64;
constexpr int kIn     = 1024;
constexpr int kHS     = 1024;
constexpr int kGates  = 4096;   // 4*HS

constexpr int kBlocksP  = 128;  // persistent grid (device-wide barrier)
constexpr int kThreadsP = 128;  // 4 wave32 per block; wave w = M-tile w

// workspace layout (all multiples of 256B)
constexpr size_t kWtBytes  = (size_t)kGates * kHS * 2;            // 8 MiB each
constexpr size_t kXbBytes  = (size_t)kSeq * kBatch * kIn * 2;     // 64 MiB
constexpr size_t kHbBytes  = (size_t)kBatch * kHS * 2;            // 128 KiB
constexpr size_t kCBytes   = (size_t)kBatch * kHS * 4;            // 256 KiB
constexpr size_t kGBytes   = (size_t)kBatch * kGates * 4;         // 1 MiB
constexpr size_t kOffWihT  = 0;
constexpr size_t kOffWhhT  = kOffWihT + kWtBytes;
constexpr size_t kOffXb    = kOffWhhT + kWtBytes;
constexpr size_t kOffHb    = kOffXb + kXbBytes;
constexpr size_t kOffC     = kOffHb + kHbBytes;
constexpr size_t kOffG     = kOffC + kCBytes;
constexpr size_t kOffBar   = kOffG + kGBytes;
}

union FragBF16 { uint4 q[2]; v16bf v; };

__device__ __forceinline__ unsigned short f32_to_bf16_rne(float f) {
  unsigned u = __float_as_uint(f);
  unsigned r = 0x7FFFu + ((u >> 16) & 1u);
  return (unsigned short)((u + r) >> 16);
}

__device__ __forceinline__ float sigmoidf_fast(float x) {
  return 1.0f / (1.0f + __expf(-x));
}

// ---- prep: f32 [R][C] -> bf16 transposed [C][R] (LDS tiled) ---------------
__global__ void k_transpose_bf16(const float* __restrict__ src,
                                 unsigned short* __restrict__ dst,
                                 int R, int C) {
  __shared__ float tile[32][33];
  const int tx = threadIdx.x & 31;
  const int ty = threadIdx.x >> 5;            // 0..7
  const int c0 = blockIdx.x * 32;
  const int r0 = blockIdx.y * 32;
#pragma unroll
  for (int i = 0; i < 32; i += 8)
    tile[ty + i][tx] = src[(size_t)(r0 + ty + i) * C + (c0 + tx)];
  __syncthreads();
#pragma unroll
  for (int i = 0; i < 32; i += 8)
    dst[(size_t)(c0 + ty + i) * R + (r0 + tx)] = f32_to_bf16_rne(tile[tx][ty + i]);
}

// ---- prep: f32 -> bf16 flat convert (float4 vectorized, grid-stride) ------
__global__ void k_f32_to_bf16(const float* __restrict__ src,
                              unsigned short* __restrict__ dst, int n) {
  const size_t stride = (size_t)gridDim.x * blockDim.x * 4;
  for (size_t i = ((size_t)blockIdx.x * blockDim.x + threadIdx.x) * 4; i < (size_t)n;
       i += stride) {
    float4 f = *(const float4*)(src + i);
    union { unsigned short s[4]; uint2 u; } o;
    o.s[0] = f32_to_bf16_rne(f.x);
    o.s[1] = f32_to_bf16_rne(f.y);
    o.s[2] = f32_to_bf16_rne(f.z);
    o.s[3] = f32_to_bf16_rne(f.w);
    *(uint2*)(dst + i) = o.u;
  }
}

// ---- prep: initialize recurrent state from h0 / c0 ------------------------
__global__ void k_init_state(const float* __restrict__ h0,
                             const float* __restrict__ c0,
                             unsigned short* __restrict__ hb,
                             float* __restrict__ cst, int n) {
  int i = blockIdx.x * blockDim.x + threadIdx.x;
  if (i < n) {
    hb[i]  = f32_to_bf16_rne(h0[i]);
    cst[i] = c0[i];
  }
}

// ---- grid-wide barrier (monotonic counter, reset by memset per launch) ----
__device__ __forceinline__ void grid_barrier(unsigned* bar, unsigned target) {
  __threadfence();                 // release our gate/h/c stores device-wide
  __syncthreads();
  if (threadIdx.x == 0) {
    __hip_atomic_fetch_add(bar, 1u, __ATOMIC_RELAXED, __HIP_MEMORY_SCOPE_AGENT);
    while (__hip_atomic_load(bar, __ATOMIC_RELAXED, __HIP_MEMORY_SCOPE_AGENT) < target)
      __builtin_amdgcn_s_sleep(1);
  }
  __syncthreads();
  __threadfence();                 // acquire others' stores
}

// ---- persistent LSTM: all 512 steps in one launch -------------------------
// Per step: gates[64,4096] = bias + x_t @ Wih + h @ Whh via v_wmma_f32_16x16x32_bf16,
// then grid barrier, then elementwise cell update, then grid barrier.
__global__ __launch_bounds__(kThreadsP) void k_lstm_persistent(
    const unsigned short* __restrict__ xb,    // [SEQ][B][IN] bf16
    const unsigned short* __restrict__ wihT,  // [GATES][IN]  bf16 (transposed)
    const unsigned short* __restrict__ whhT,  // [GATES][HS]  bf16 (transposed)
    const float* __restrict__ bias,           // [GATES]
    unsigned short* __restrict__ hb,          // [B][HS] bf16 recurrent state
    float* __restrict__ cst,                  // [B][HS] f32 cell state
    float* __restrict__ gates,                // [B][GATES] f32 scratch
    unsigned* __restrict__ bar,               // barrier counter (zeroed)
    float* __restrict__ out)                  // hidden_seq | h_T | c_T
{
  const int tid   = threadIdx.x;
  const int lane  = tid & 31;
  const int mt    = tid >> 5;                 // wave index == M tile (0..3)
  const int l15   = lane & 15;
  const int khalf = lane >> 4;                // 0..1
  // lane's two output columns (C layout: N = lane&15 within each 16-wide tile)
  const int ncol0 = blockIdx.x * 32 + l15;
  const int ncol1 = ncol0 + 16;
  // lane's A row (A layout: M = lane&15)
  const int arow  = mt * 16 + l15;

  const float bval0 = bias[ncol0];
  const float bval1 = bias[ncol1];
  const unsigned short* __restrict__ wih0 = wihT + (size_t)ncol0 * kIn;
  const unsigned short* __restrict__ wih1 = wihT + (size_t)ncol1 * kIn;
  const unsigned short* __restrict__ whh0 = whhT + (size_t)ncol0 * kHS;
  const unsigned short* __restrict__ whh1 = whhT + (size_t)ncol1 * kHS;

  unsigned barTarget = 0;

  for (int t = 0; t < kSeq; ++t) {
    v8f acc0, acc1;
#pragma unroll
    for (int v = 0; v < 8; ++v) { acc0[v] = bval0; acc1[v] = bval1; }

    // ---- gate tiles = x_t @ Wih (K = 1024); A shared by both N-tiles ----
    const unsigned short* __restrict__ xa =
        xb + (size_t)t * (kBatch * kIn) + (size_t)arow * kIn;
#pragma unroll 4
    for (int kb = 0; kb < kIn; kb += 32) {
      FragBF16 a, b0, b1;
      // A 16x32 bf16: lane holds row `arow`; K = kb+khalf*16 .. +16
      a.q[0] = *(const uint4*)(xa + kb + khalf * 16);
      a.q[1] = *(const uint4*)(xa + kb + khalf * 16 + 8);
      // B 32x16 bf16 (transposed weights): lane holds one column, K = kb+khalf*16 .. +16
      const unsigned short* bp0 = wih0 + kb + khalf * 16;
      const unsigned short* bp1 = wih1 + kb + khalf * 16;
      b0.q[0] = *(const uint4*)(bp0);
      b0.q[1] = *(const uint4*)(bp0 + 8);
      b1.q[0] = *(const uint4*)(bp1);
      b1.q[1] = *(const uint4*)(bp1 + 8);
      acc0 = __builtin_amdgcn_wmma_f32_16x16x32_bf16(
          false, a.v, false, b0.v, (short)0, acc0, false, false);
      acc1 = __builtin_amdgcn_wmma_f32_16x16x32_bf16(
          false, a.v, false, b1.v, (short)0, acc1, false, false);
    }

    // ---- += h @ Whh (K = 1024) ----
    const unsigned short* __restrict__ ha = hb + (size_t)arow * kHS;
#pragma unroll 4
    for (int kb = 0; kb < kHS; kb += 32) {
      FragBF16 a, b0, b1;
      a.q[0] = *(const uint4*)(ha + kb + khalf * 16);
      a.q[1] = *(const uint4*)(ha + kb + khalf * 16 + 8);
      const unsigned short* bp0 = whh0 + kb + khalf * 16;
      const unsigned short* bp1 = whh1 + kb + khalf * 16;
      b0.q[0] = *(const uint4*)(bp0);
      b0.q[1] = *(const uint4*)(bp0 + 8);
      b1.q[0] = *(const uint4*)(bp1);
      b1.q[1] = *(const uint4*)(bp1 + 8);
      acc0 = __builtin_amdgcn_wmma_f32_16x16x32_bf16(
          false, a.v, false, b0.v, (short)0, acc0, false, false);
      acc1 = __builtin_amdgcn_wmma_f32_16x16x32_bf16(
          false, a.v, false, b1.v, (short)0, acc1, false, false);
    }

    // ---- store gate tiles (C layout: VGPR v -> row mt*16 + khalf*8 + v) ----
    {
      float* gp = gates + (size_t)(mt * 16 + khalf * 8) * kGates + ncol0;
#pragma unroll
      for (int v = 0; v < 8; ++v) {
        gp[(size_t)v * kGates]      = acc0[v];
        gp[(size_t)v * kGates + 16] = acc1[v];
      }
    }

    barTarget += gridDim.x;
    grid_barrier(bar, barTarget);

    // ---- elementwise LSTM cell update: 64*1024 cells over 16384 threads ----
    for (int idx = blockIdx.x * kThreadsP + tid; idx < kBatch * kHS;
         idx += kBlocksP * kThreadsP) {
      const int bi = idx >> 10;          // /HS
      const int j  = idx & (kHS - 1);
      const float* grow = gates + (size_t)bi * kGates;
      const float ig = grow[j];
      const float fg = grow[kHS + j];
      const float gg = grow[2 * kHS + j];
      const float og = grow[3 * kHS + j];
      const float cn = sigmoidf_fast(fg) * cst[idx] + sigmoidf_fast(ig) * tanhf(gg);
      const float hn = sigmoidf_fast(og) * tanhf(cn);
      cst[idx] = cn;
      hb[idx]  = f32_to_bf16_rne(hn);
      out[(size_t)t * (kBatch * kHS) + idx] = hn;
      if (t == kSeq - 1) {
        out[(size_t)kSeq * kBatch * kHS + idx] = hn;                  // h_T
        out[(size_t)kSeq * kBatch * kHS + kBatch * kHS + idx] = cn;   // c_T
      }
    }

    barTarget += gridDim.x;
    grid_barrier(bar, barTarget);
  }
}

// ---------------------------------------------------------------------------
extern "C" void kernel_launch(void* const* d_in, const int* in_sizes, int n_in,
                              void* d_out, int out_size, void* d_ws, size_t ws_size,
                              hipStream_t stream) {
  (void)in_sizes; (void)n_in; (void)out_size; (void)ws_size;
  const float* x    = (const float*)d_in[0];   // [512,64,1024]
  const float* h0   = (const float*)d_in[1];   // [1,64,1024]
  const float* c0   = (const float*)d_in[2];   // [1,64,1024]
  const float* wih  = (const float*)d_in[3];   // [1024,4096]
  const float* whh  = (const float*)d_in[4];   // [1024,4096]
  const float* bias = (const float*)d_in[5];   // [4096]
  float* out = (float*)d_out;

  char* ws = (char*)d_ws;
  unsigned short* wihT = (unsigned short*)(ws + kOffWihT);
  unsigned short* whhT = (unsigned short*)(ws + kOffWhhT);
  unsigned short* xb   = (unsigned short*)(ws + kOffXb);
  unsigned short* hb   = (unsigned short*)(ws + kOffHb);
  float*          cst  = (float*)(ws + kOffC);
  float*          gbuf = (float*)(ws + kOffG);
  unsigned*       bar  = (unsigned*)(ws + kOffBar);

  // barrier counter must start at 0 every call (graph-capture safe memset node)
  hipMemsetAsync(bar, 0, 256, stream);

  // weights: f32 [K][4H] -> bf16 [4H][K]
  k_transpose_bf16<<<dim3(kGates / 32, kIn / 32), 256, 0, stream>>>(wih, wihT, kIn, kGates);
  k_transpose_bf16<<<dim3(kGates / 32, kHS / 32), 256, 0, stream>>>(whh, whhT, kHS, kGates);

  // x -> bf16
  k_f32_to_bf16<<<8192, 256, 0, stream>>>(x, xb, kSeq * kBatch * kIn);

  // h0/c0 -> state buffers
  k_init_state<<<(kBatch * kHS + 255) / 256, 256, 0, stream>>>(h0, c0, hb, cst,
                                                               kBatch * kHS);

  // persistent recurrent kernel: 128 blocks x 128 threads = 512 wave32,
  // each wave computes two 16x16 gate tiles per step.
  k_lstm_persistent<<<kBlocksP, kThreadsP, 0, stream>>>(xb, wihT, whhT, bias, hb, cst,
                                                        gbuf, bar, out);
}
